// NT_Xent_SingGPU_83588653515237
// MI455X (gfx1250) — matrix operand order repaired and checked by reading, not compiled, not run on hardware
//
#include <hip/hip_runtime.h>
#include <hip/hip_bf16.h>
#include <hip/hip_fp16.h>

// NT-Xent (SimCLR) fused loss for MI455X / gfx1250, wave32 + WMMA.
//
// N = 8192 rows, D = 128. sim = zn @ zn^T * 2 is 17.2 GFLOP; inputs are only
// 4 MB (L2/WGP$-resident), so never materialize the 256 MB sim matrix:
// compute 16x16 tiles with v_wmma_f32_16x16x32_f16 (split-f16 hi/lo for
// near-fp32 accuracy: hi*hi + hi*lo + lo*hi, f32 accumulate) and fuse the
// exp/logsumexp + comparison counting in registers.
//
// Epilogue notes:
//  * diagonal / positive-pair masking can only occur on 2 of the 512 column
//    tiles (ct == rt and ct == rt^256). rt is forced into an SGPR via
//    readfirstlane so this check is a scalar s_cmp/s_cbranch (EXEC untouched),
//    not an if-converted exec-mask dance.
//  * exp uses __builtin_amdgcn_exp2f (raw v_exp_f32) -- the libm exp2f range
//    guard (cndmask/ldexp for args < -126) is dead weight for args in [-6, 0].

typedef _Float16 half8  __attribute__((ext_vector_type(8)));
typedef _Float16 half4v __attribute__((ext_vector_type(4)));
typedef _Float16 v16h   __attribute__((ext_vector_type(16)));
typedef float    v8f    __attribute__((ext_vector_type(8)));

constexpr int   kB       = 4096;   // batch
constexpr int   kN       = 8192;   // 2*batch
constexpr int   kD       = 128;    // dim
constexpr float kInvTemp = 2.0f;   // 1 / 0.5
// exp(2*acc - 2) == exp2(acc * 2*log2(e) - 2*log2(e))
constexpr float kExpScale = 2.0f * 1.44269504088896340736f;

// ---------------------------------------------------------------------------
// Kernel 1: row-normalize concat(z_i, z_j); emit f16 hi and f16 lo planes.
// One wave per row (32 lanes x float4 = 128 elems).
// ---------------------------------------------------------------------------
__global__ void __launch_bounds__(256) norm_split_kernel(
    const float* __restrict__ zi, const float* __restrict__ zj,
    _Float16* __restrict__ zh, _Float16* __restrict__ zl) {
  const int row  = blockIdx.x * 8 + (threadIdx.x >> 5);
  const int lane = threadIdx.x & 31;
  const float* src = (row < kB) ? (zi + (size_t)row * kD)
                                : (zj + (size_t)(row - kB) * kD);
  float4 v = ((const float4*)src)[lane];
  float ss = v.x * v.x + v.y * v.y + v.z * v.z + v.w * v.w;
#pragma unroll
  for (int m = 16; m >= 1; m >>= 1) ss += __shfl_xor(ss, m, 32);
  const float inv = 1.0f / fmaxf(sqrtf(ss), 1e-8f);
  float a[4] = {v.x * inv, v.y * inv, v.z * inv, v.w * inv};
  half4v hv, lv;
#pragma unroll
  for (int j = 0; j < 4; ++j) {
    _Float16 hh = (_Float16)a[j];
    hv[j] = hh;
    lv[j] = (_Float16)(a[j] - (float)hh);
  }
  ((half4v*)(zh + (size_t)row * kD))[lane] = hv;
  ((half4v*)(zl + (size_t)row * kD))[lane] = lv;
}

// ---------------------------------------------------------------------------
// Kernel 2: positive logits pos[i] = dot(zn[i], zn[i^B]) * (1/TEMP).
// Reconstruct zn = hi + lo (error ~2^-22). One wave per row.
// ---------------------------------------------------------------------------
__global__ void __launch_bounds__(256) pos_kernel(
    const _Float16* __restrict__ zh, const _Float16* __restrict__ zl,
    float* __restrict__ pos) {
  const int row  = blockIdx.x * 8 + (threadIdx.x >> 5);
  const int lane = threadIdx.x & 31;
  const int pr   = row ^ kB;  // (row + B) mod N, since B is a power of two
  half4v ah = ((const half4v*)(zh + (size_t)row * kD))[lane];
  half4v al = ((const half4v*)(zl + (size_t)row * kD))[lane];
  half4v bh = ((const half4v*)(zh + (size_t)pr  * kD))[lane];
  half4v bl = ((const half4v*)(zl + (size_t)pr  * kD))[lane];
  float d = 0.0f;
#pragma unroll
  for (int j = 0; j < 4; ++j)
    d += ((float)ah[j] + (float)al[j]) * ((float)bh[j] + (float)bl[j]);
#pragma unroll
  for (int m = 16; m >= 1; m >>= 1) d += __shfl_xor(d, m, 32);
  if (lane == 0) pos[row] = d * kInvTemp;
}

// ---------------------------------------------------------------------------
// Kernel 3: fused sim-tile GEMM + logsumexp-sum + comparison count.
// Each wave owns rows [16*rt, 16*rt+16) and sweeps all 512 column tiles.
// A/B lane layouts per CDNA5 ISA 7.12.2 (16-bit A 16x32, B 32x16, f32 C/D).
// sim <= 2, so exp(sim-2) is a safe fixed-shift sumexp (no online max).
// ---------------------------------------------------------------------------
__global__ void __launch_bounds__(128) sim_reduce_kernel(
    const _Float16* __restrict__ zh, const _Float16* __restrict__ zl,
    const float* __restrict__ pos,
    float* __restrict__ Ssum, float* __restrict__ Csum) {
  const int wid  = threadIdx.x >> 5;
  // Force the (actually wave-uniform) row-tile index into an SGPR so every
  // dependent comparison/branch is scalar.
  const int rt   = __builtin_amdgcn_readfirstlane(blockIdx.x * 4 + wid);
  const int lane = threadIdx.x & 31;
  const int h    = lane >> 4;   // half-of-wave id
  const int q    = lane & 15;   // M for A-frags, N for B/C/D-frags

  // Hoist A fragments: 4 K-chunks of 32, hi and lo planes.
  // Lane layout: a[j]   = A[m][kc + h*8 + j]       (j = 0..7)
  //              a[8+j] = A[m][kc + h*8 + 16 + j]
  v16h Ah[4], Al[4];
  {
    const size_t arow = (size_t)(rt * 16 + q) * kD;
#pragma unroll
    for (int c = 0; c < 4; ++c) {
      const _Float16* pa = zh + arow + c * 32 + h * 8;
      const _Float16* pl = zl + arow + c * 32 + h * 8;
      half8 a0 = *(const half8*)(pa);
      half8 a1 = *(const half8*)(pa + 16);
      half8 l0 = *(const half8*)(pl);
      half8 l1 = *(const half8*)(pl + 16);
#pragma unroll
      for (int j = 0; j < 8; ++j) {
        Ah[c][j] = a0[j]; Ah[c][8 + j] = a1[j];
        Al[c][j] = l0[j]; Al[c][8 + j] = l1[j];
      }
    }
  }

  // posh = pos/2 so the negative test "pos > sim" becomes "posh > acc"
  // (acc is the un-scaled cosine accumulator) -- no per-element scaling.
  float posh[8];
#pragma unroll
  for (int r = 0; r < 8; ++r) posh[r] = 0.5f * pos[rt * 16 + h * 8 + r];

  float sacc[8] = {0, 0, 0, 0, 0, 0, 0, 0};
  float cacc[8] = {0, 0, 0, 0, 0, 0, 0, 0};

  const int pair_rt = rt ^ (kB / 16);  // only tile where a positive pair hits

  for (int ct = 0; ct < kN / 16; ++ct) {
    const size_t brow = (size_t)(ct * 16 + q) * kD;
    // Prefetch next tile's B rows (global_prefetch_b8).
    __builtin_prefetch(zh + (size_t)((((ct + 1) & 511) * 16) + q) * kD, 0, 0);

    v8f acc = {0.f, 0.f, 0.f, 0.f, 0.f, 0.f, 0.f, 0.f};
#pragma unroll
    for (int c = 0; c < 4; ++c) {
      // B lane layout: b[j] = B[kc + h*8 + j][n], b[8+j] = B[kc + h*8 + 16 + j][n].
      const _Float16* pb  = zh + brow + c * 32 + h * 8;
      const _Float16* pbl = zl + brow + c * 32 + h * 8;
      half8 b0  = *(const half8*)(pb);
      half8 b1  = *(const half8*)(pb + 16);
      half8 bl0 = *(const half8*)(pbl);
      half8 bl1 = *(const half8*)(pbl + 16);
      v16h Bh, Bl;
#pragma unroll
      for (int j = 0; j < 8; ++j) {
        Bh[j] = b0[j];  Bh[8 + j] = b1[j];
        Bl[j] = bl0[j]; Bl[8 + j] = bl1[j];
      }
      // sim ~= hi*hi + hi*lo + lo*hi  (split-f16, f32 accumulate)
      acc = __builtin_amdgcn_wmma_f32_16x16x32_f16(false, Ah[c], false, Bh,
                                                   (short)0, acc, false, false);
      acc = __builtin_amdgcn_wmma_f32_16x16x32_f16(false, Ah[c], false, Bl,
                                                   (short)0, acc, false, false);
      acc = __builtin_amdgcn_wmma_f32_16x16x32_f16(false, Al[c], false, Bh,
                                                   (short)0, acc, false, false);
    }

    // Epilogue. Diagonal hits only at ct==rt; positive pair only at
    // ct==rt^256. Both operands are SGPRs -> s_cmp + s_cbranch, no
    // divergence, EXEC stays all-ones on both paths.
    if (__builtin_expect((ct == rt) || (ct == pair_rt), 0)) {
      const int gj = ct * 16 + q;
#pragma unroll
      for (int r = 0; r < 8; ++r) {
        const int gi    = rt * 16 + h * 8 + r;
        const float e   =
            __builtin_amdgcn_exp2f(fmaf(acc[r], kExpScale, -kExpScale));
        const bool diag = (gi == gj);
        sacc[r] += diag ? 0.0f : e;                    // LSE excl. diagonal
        const bool neg = (!diag) & ((gi ^ gj) != kB);  // not diag, not pair
        cacc[r] += (neg & (posh[r] > acc[r])) ? 1.0f : 0.0f;
      }
    } else {
      // Fast path (510/512 tiles): every element is a negative.
#pragma unroll
      for (int r = 0; r < 8; ++r) {
        sacc[r] +=
            __builtin_amdgcn_exp2f(fmaf(acc[r], kExpScale, -kExpScale));
        cacc[r] += (posh[r] > acc[r]) ? 1.0f : 0.0f;
      }
    }
  }

  // Reduce across the 16 lanes sharing each output row; wave owns its rows.
#pragma unroll
  for (int r = 0; r < 8; ++r) {
    float s = sacc[r], cc = cacc[r];
#pragma unroll
    for (int m = 8; m >= 1; m >>= 1) {
      s  += __shfl_xor(s, m, 32);
      cc += __shfl_xor(cc, m, 32);
    }
    if (q == 0) {
      const int gi = rt * 16 + h * 8 + r;
      Ssum[gi] = s;
      Csum[gi] = cc;
    }
  }
}

// ---------------------------------------------------------------------------
// Kernel 4: final scalar reduction -> d_out = {loss, acc}.
// ---------------------------------------------------------------------------
__global__ void __launch_bounds__(256) finalize_kernel(
    const float* __restrict__ Ssum, const float* __restrict__ Csum,
    const float* __restrict__ pos, float* __restrict__ out) {
  __shared__ float sl[256];
  __shared__ float sc[256];
  const int tid = threadIdx.x;
  float ll = 0.0f, cc = 0.0f;
  for (int i = tid; i < kN; i += 256) {
    // lse_i = log(sum exp(sim - 2)) + 2 ; loss term = lse_i - pos_i
    ll += logf(Ssum[i]) + 2.0f - pos[i];
    cc += Csum[i];
  }
  sl[tid] = ll;
  sc[tid] = cc;
  __syncthreads();
  for (int s = 128; s > 0; s >>= 1) {
    if (tid < s) { sl[tid] += sl[tid + s]; sc[tid] += sc[tid + s]; }
    __syncthreads();
  }
  if (tid == 0) {
    out[0] = sl[0] / (float)kN;
    out[1] = sc[0] / (float)kN / (float)kN;
  }
}

// ---------------------------------------------------------------------------
extern "C" void kernel_launch(void* const* d_in, const int* in_sizes, int n_in,
                              void* d_out, int out_size, void* d_ws,
                              size_t ws_size, hipStream_t stream) {
  (void)in_sizes; (void)n_in; (void)out_size; (void)ws_size;
  const float* zi = (const float*)d_in[0];
  const float* zj = (const float*)d_in[1];

  // Workspace layout (~4.1 MB total):
  char* ws = (char*)d_ws;
  _Float16* zh  = (_Float16*)(ws);                    // 8192*128 f16 = 2 MB
  _Float16* zl  = (_Float16*)(ws + (2u << 20));       // 2 MB
  float*    pos = (float*)(ws + (4u << 20));          // 32 KB
  float*    S   = pos + kN;                           // 32 KB
  float*    C   = S + kN;                             // 32 KB
  float*    out = (float*)d_out;

  norm_split_kernel<<<kN / 8, 256, 0, stream>>>(zi, zj, zh, zl);
  pos_kernel<<<kN / 8, 256, 0, stream>>>(zh, zl, pos);
  sim_reduce_kernel<<<(kN / 16) / 4, 128, 0, stream>>>(zh, zl, pos, S, C);
  finalize_kernel<<<1, 256, 0, stream>>>(S, C, pos, out);
}
